// MultiHeadAttention_35038343201225
// MI455X (gfx1250) — compile-verified, matrix operand order checked by reading, not run on hardware
//
#include <hip/hip_runtime.h>
#include <hip/hip_bf16.h>

typedef __bf16 bf16_t;
typedef __attribute__((ext_vector_type(16))) __bf16 v16bf;
typedef __attribute__((ext_vector_type(8)))  __bf16 v8bf;
typedef __attribute__((ext_vector_type(8)))  float   v8f;

#define SEQ    2048
#define DMODEL 1024
#define NHEAD  16
#define HDIM   64
#define ROWS   4096   // b * n
#define KDIM   1024
#define NSTEP  (KDIM / 32)

#if defined(__HIP_DEVICE_COMPILE__) && defined(__gfx1250__)
#define ASYNC_LDS 1
#else
#define ASYNC_LDS 0
#endif

// ---------------------------------------------------------------------------
// Fast device helpers
// ---------------------------------------------------------------------------
__device__ __forceinline__ float fast_exp2(float x) {
#if defined(__HIP_DEVICE_COMPILE__)
  return __builtin_amdgcn_exp2f(x);   // raw v_exp_f32 (flush below -126: fine
#else                                 // for softmax numerators)
  return exp2f(x);
#endif
}

__device__ __forceinline__ float lane_bcast(float v, int byte_idx) {
#if defined(__HIP_DEVICE_COMPILE__)
  return __int_as_float(__builtin_amdgcn_ds_bpermute(byte_idx, __float_as_int(v)));
#else
  return v;
#endif
}

// ---------------------------------------------------------------------------
// Fragment loader: row-major [rows x ld] bf16 source.
// CDNA5 16-bit A/B 16x32 layout: lane group g (lane>>4) holds
//   elements 0..7  -> K = kbase + g*8 + i
//   elements 8..15 -> K = kbase + 16 + g*8 + i
// ---------------------------------------------------------------------------
__device__ __forceinline__ v16bf load_frag(const bf16_t* __restrict__ base,
                                           int row, int ld, int kbase, int g) {
  const bf16_t* p = base + (size_t)row * (size_t)ld + (size_t)(kbase + g * 8);
  v8bf lo = *(const v8bf*)(p);
  v8bf hi = *(const v8bf*)(p + 16);
  v16bf r;
#pragma unroll
  for (int i = 0; i < 8; ++i) { r[i] = lo[i]; r[i + 8] = hi[i]; }
  return r;
}

__device__ __forceinline__ v8f wmma_bf16(v16bf a, v16bf b, v8f c) {
  return __builtin_amdgcn_wmma_f32_16x16x32_bf16(false, a, false, b,
                                                 (short)0, c, false, false);
}

// Async global->LDS copy of 16 bf16 elements (two b128 per lane).
// ISA 15.18.3 op 98: GLOBAL_LOAD_ASYNC_TO_LDS_B128; VDST = LDS byte address,
// VADDR = 64-bit global address, saddr = off. Tracked by ASYNCcnt.
__device__ __forceinline__ void stage_pair(bf16_t* lds, const bf16_t* g) {
#if ASYNC_LDS
  unsigned off0 = (unsigned)(size_t)(const void*)lds;
  asm volatile("global_load_async_to_lds_b128 %0, %1, off"
               :: "v"(off0), "v"(g) : "memory");
  unsigned off1 = off0 + 16;
  asm volatile("global_load_async_to_lds_b128 %0, %1, off"
               :: "v"(off1), "v"(g + 8) : "memory");
#else
  *(v8bf*)lds       = *(const v8bf*)g;
  *(v8bf*)(lds + 8) = *(const v8bf*)(g + 8);
#endif
}

template <int N>
__device__ __forceinline__ void wait_async() {
#if ASYNC_LDS
  asm volatile("s_wait_asynccnt %0" :: "n"(N));
#endif
}

// ---------------------------------------------------------------------------
// f32 -> bf16 cast, 8 elements per thread (b128 in / b128 out)
// ---------------------------------------------------------------------------
__global__ void cast_f32_to_bf16(const float* __restrict__ in,
                                 bf16_t* __restrict__ out, int n8) {
  int i = blockIdx.x * blockDim.x + threadIdx.x;
  if (i >= n8) return;
  int base = i * 8;
  float4 x0 = *(const float4*)(in + base);
  float4 x1 = *(const float4*)(in + base + 4);
  v8bf o;
  o[0] = (bf16_t)x0.x; o[1] = (bf16_t)x0.y; o[2] = (bf16_t)x0.z; o[3] = (bf16_t)x0.w;
  o[4] = (bf16_t)x1.x; o[5] = (bf16_t)x1.y; o[6] = (bf16_t)x1.z; o[7] = (bf16_t)x1.w;
  *(v8bf*)(out + base) = o;
}

// ---------------------------------------------------------------------------
// GEMM: out[4096 x 1024] = A[4096 x 1024](bf16) * W[1024 x 1024]^T(bf16)
//       + bias, then * scale.
// Block = 4 waves; each wave owns a 32x64 tile (8 WMMAs / k-step). The block
// shares a 64x32 W tile per k-step, staged in LDS by async loads, 4-deep
// buffered (index = i & 3), one barrier per step. A fragments ping-pong
// between two register sets (loop unrolled by 2, no v_movs).
// MODE: 0 = f32 row-major, 1 = bf16 row-major,
//       2 = bf16 per-batch transposed ([batch][col][n]) for V^T.
// ---------------------------------------------------------------------------
template <int MODE>
__global__ void __launch_bounds__(128)
gemm_xwT(const bf16_t* __restrict__ A, const bf16_t* __restrict__ W,
         const float* __restrict__ bias, void* __restrict__ out, float scale) {
  __shared__ bf16_t shB[4 * 2048];   // 4 buffers x (64 rows x 32 cols) bf16

  const int tid  = threadIdx.x;
  const int w    = tid >> 5;
  const int lane = tid & 31;
  const int g    = lane >> 4;
  const int r16  = lane & 15;

  const int pm  = blockIdx.x >> 4;           // 128-row panel (N/64 = 16)
  const int ng  = blockIdx.x & 15;
  const int rowbase = pm * 128 + w * 32;

  // Staging role: 128 lanes cover 64 rows x 2 half-chunks of 16 elements.
  const int srow  = tid >> 1;
  const int shalf = tid & 1;
  const bf16_t* gsrc = W + (size_t)(ng * 64 + srow) * KDIM + shalf * 16;
  bf16_t*       lsta = &shB[0] + srow * 32 + shalf * 16;

  v8f acc[2][4] = {};

  // one k-step: compute on (ca0, ca1) for step i; stage buffer i+1 and load
  // the next A fragments into (na0, na1).
  auto step_staged = [&](int i, v16bf& ca0, v16bf& ca1,
                         v16bf& na0, v16bf& na1) {
    const int k = i << 5;
    stage_pair(lsta + ((i + 1) & 3) * 2048, gsrc + (size_t)(k + 32));
    wait_async<2>();                 // the 2 asyncs for buffer i are done
    __syncthreads();
    __builtin_prefetch(A + (size_t)(rowbase + r16) * KDIM + k + 64 + g * 8, 0, 3);
    __builtin_prefetch(A + (size_t)(rowbase + 16 + r16) * KDIM + k + 64 + g * 8, 0, 3);
    na0 = load_frag(A, rowbase + r16,      KDIM, k + 32, g);
    na1 = load_frag(A, rowbase + 16 + r16, KDIM, k + 32, g);
    const bf16_t* bbuf = &shB[(i & 3) * 2048];
#pragma unroll
    for (int t = 0; t < 4; ++t) {
      v16bf b = load_frag(bbuf, t * 16 + r16, 32, 0, g);
      acc[0][t] = wmma_bf16(ca0, b, acc[0][t]);
      acc[1][t] = wmma_bf16(ca1, b, acc[1][t]);
    }
  };

  // Prologue: stage k=0 into buffer 0; load A fragments for k=0.
  stage_pair(lsta, gsrc);
  v16bf aA0 = load_frag(A, rowbase + r16,      KDIM, 0, g);
  v16bf aA1 = load_frag(A, rowbase + 16 + r16, KDIM, 0, g);
  v16bf aB0, aB1;

  for (int i = 0; i < NSTEP - 2; i += 2) {   // steady state: branch-free body
    step_staged(i,     aA0, aA1, aB0, aB1);
    step_staged(i + 1, aB0, aB1, aA0, aA1);
  }
  step_staged(NSTEP - 2, aA0, aA1, aB0, aB1);  // stages buffer for step 31

  {                                            // last step: no staging
    const int i = NSTEP - 1;
    wait_async<0>();
    __syncthreads();
    const bf16_t* bbuf = &shB[(i & 3) * 2048];
#pragma unroll
    for (int t = 0; t < 4; ++t) {
      v16bf b = load_frag(bbuf, t * 16 + r16, 32, 0, g);
      acc[0][t] = wmma_bf16(aB0, b, acc[0][t]);
      acc[1][t] = wmma_bf16(aB1, b, acc[1][t]);
    }
  }

#pragma unroll
  for (int s = 0; s < 2; ++s) {
#pragma unroll
    for (int t = 0; t < 4; ++t) {
      int col = ng * 64 + t * 16 + r16;
      float bv = bias[col];
#pragma unroll
      for (int j = 0; j < 8; ++j) {
        int rr = rowbase + s * 16 + g * 8 + j;   // C/D: M = g*8+j
        float val = (acc[s][t][j] + bv) * scale;
        if constexpr (MODE == 0) {
          ((float*)out)[(size_t)rr * DMODEL + col] = val;
        } else if constexpr (MODE == 1) {
          ((bf16_t*)out)[(size_t)rr * DMODEL + col] = (bf16_t)val;
        } else {
          int batch = rr >> 11;
          int nn    = rr & (SEQ - 1);
          ((bf16_t*)out)[((size_t)batch * DMODEL + col) * SEQ + nn] = (bf16_t)val;
        }
      }
    }
  }
}

// ---------------------------------------------------------------------------
// Flash-style attention: one wave per (b, h, 16-row q-tile).
// Computes S^T = K_block * Q^T so the S^T C-fragment directly reinterprets
// as the A-fragment for P @ V. Q is pre-scaled by log2(e)/sqrt(d), so the
// softmax runs natively in base-2 with raw v_exp_f32.
// ---------------------------------------------------------------------------
__global__ void __launch_bounds__(128)
attn_kernel(const bf16_t* __restrict__ Q, const bf16_t* __restrict__ Kd,
            const bf16_t* __restrict__ Vt, bf16_t* __restrict__ Oout) {
  int wid  = (blockIdx.x * blockDim.x + threadIdx.x) >> 5;
  int lane = threadIdx.x & 31;
  int g    = lane >> 4;
  int r16  = lane & 15;

  int qt = wid & 127;              // SEQ/16 q-tiles
  int bh = wid >> 7;
  int b  = bh >> 4;
  int h  = bh & 15;

  const bf16_t* Qb = Q  + ((size_t)b * SEQ) * DMODEL + (size_t)h * HDIM;
  const bf16_t* Kb = Kd + ((size_t)b * SEQ) * DMODEL + (size_t)h * HDIM;
  const bf16_t* Vb = Vt + ((size_t)b * DMODEL + (size_t)h * HDIM) * SEQ;

  // Loop-invariant bpermute byte indices: row j of an O tile needs the scalar
  // of q = g*8 + j, which lives at lane (g*8 + j).
  int bpidx[8];
#pragma unroll
  for (int j = 0; j < 8; ++j) bpidx[j] = ((g << 3) + j) << 2;

  int qrow = qt * 16 + r16;
  v16bf qf0 = load_frag(Qb, qrow, DMODEL, 0,  g);
  v16bf qf1 = load_frag(Qb, qrow, DMODEL, 32, g);

  float m = -1e30f, l = 0.0f;
  v8f o[4] = {v8f{}, v8f{}, v8f{}, v8f{}};
  const v8f zero = {};

  for (int kb = 0; kb < SEQ; kb += 32) {
    if (kb + 32 < SEQ) {             // prefetch next K/V block
      __builtin_prefetch(Kb + (size_t)(kb + 32 + r16) * DMODEL + g * 8, 0, 3);
      __builtin_prefetch(Kb + (size_t)(kb + 48 + r16) * DMODEL + g * 8, 0, 3);
#pragma unroll
      for (int t = 0; t < 4; ++t)
        __builtin_prefetch(Vb + (size_t)(t * 16 + r16) * SEQ + kb + 32 + g * 8, 0, 3);
    }

    v16bf ka0 = load_frag(Kb, kb + r16, DMODEL, 0,  g);
    v16bf ka1 = load_frag(Kb, kb + r16, DMODEL, 32, g);
    v8f s0 = wmma_bf16(ka0, qf0, zero);
    s0 = wmma_bf16(ka1, qf1, s0);
    v16bf kc0 = load_frag(Kb, kb + 16 + r16, DMODEL, 0,  g);
    v16bf kc1 = load_frag(Kb, kb + 16 + r16, DMODEL, 32, g);
    v8f s1 = wmma_bf16(kc0, qf0, zero);
    s1 = wmma_bf16(kc1, qf1, s1);

    // Per-lane: 16 k-values for q = r16; complement lives at lane ^ 16.
    float bm = s0[0];
#pragma unroll
    for (int j = 0; j < 8; ++j) { bm = fmaxf(bm, s0[j]); bm = fmaxf(bm, s1[j]); }
    bm = fmaxf(bm, __shfl_xor(bm, 16));

    float mn   = fmaxf(m, bm);
    float corr = fast_exp2(m - mn);  // base-2 softmax, raw v_exp_f32
    float rs   = 0.0f;
    v16bf pf;
#pragma unroll
    for (int j = 0; j < 8; ++j) {
      float p0 = fast_exp2(s0[j] - mn);
      float p1 = fast_exp2(s1[j] - mn);
      rs += p0 + p1;
      pf[j]     = (bf16_t)p0;        // S^T tile0 C-values == A-frag elems 0..7
      pf[j + 8] = (bf16_t)p1;        // S^T tile1 C-values == A-frag elems 8..15
    }
    rs += __shfl_xor(rs, 16);
    l = l * corr + rs;
    m = mn;

    float cr[8];
#pragma unroll
    for (int j = 0; j < 8; ++j) cr[j] = lane_bcast(corr, bpidx[j]);
#pragma unroll
    for (int t = 0; t < 4; ++t) {
#pragma unroll
      for (int j = 0; j < 8; ++j) o[t][j] *= cr[j];
    }

#pragma unroll
    for (int t = 0; t < 4; ++t) {
      v16bf vf = load_frag(Vb, t * 16 + r16, SEQ, kb, g);
      o[t] = wmma_bf16(pf, vf, o[t]);
    }
  }

  float linv = 1.0f / l;
  float lr[8];
#pragma unroll
  for (int j = 0; j < 8; ++j) lr[j] = lane_bcast(linv, bpidx[j]);

#pragma unroll
  for (int t = 0; t < 4; ++t) {
#pragma unroll
    for (int j = 0; j < 8; ++j) {
      int rr  = qt * 16 + g * 8 + j;
      int col = h * HDIM + t * 16 + r16;
      Oout[((size_t)b * SEQ + rr) * DMODEL + col] = (bf16_t)(o[t][j] * lr[j]);
    }
  }
}

// ---------------------------------------------------------------------------
// Launcher
// ---------------------------------------------------------------------------
extern "C" void kernel_launch(void* const* d_in, const int* in_sizes, int n_in,
                              void* d_out, int out_size, void* d_ws, size_t ws_size,
                              hipStream_t stream) {
  const float* x  = (const float*)d_in[0];
  const float* Wq = (const float*)d_in[1];
  const float* bq = (const float*)d_in[2];
  const float* Wk = (const float*)d_in[3];
  const float* bk = (const float*)d_in[4];
  const float* Wv = (const float*)d_in[5];
  const float* bv = (const float*)d_in[6];
  const float* Wo = (const float*)d_in[7];
  const float* bo = (const float*)d_in[8];

  size_t off = 0;
  auto wsalloc = [&](size_t bytes) -> void* {
    void* p = (char*)d_ws + off;
    off += (bytes + 255) & ~(size_t)255;
    return p;
  };
  bf16_t* xbf   = (bf16_t*)wsalloc((size_t)ROWS * DMODEL * 2);
  bf16_t* wqbf  = (bf16_t*)wsalloc((size_t)DMODEL * DMODEL * 2);
  bf16_t* wkbf  = (bf16_t*)wsalloc((size_t)DMODEL * DMODEL * 2);
  bf16_t* wvbf  = (bf16_t*)wsalloc((size_t)DMODEL * DMODEL * 2);
  bf16_t* wobf  = (bf16_t*)wsalloc((size_t)DMODEL * DMODEL * 2);
  bf16_t* qbf   = (bf16_t*)wsalloc((size_t)ROWS * DMODEL * 2);
  bf16_t* kbf   = (bf16_t*)wsalloc((size_t)ROWS * DMODEL * 2);
  bf16_t* vtbf  = (bf16_t*)wsalloc((size_t)ROWS * DMODEL * 2);  // [b][d][n]
  bf16_t* attnb = (bf16_t*)wsalloc((size_t)ROWS * DMODEL * 2);

  const int nx8 = ROWS * DMODEL / 8;     // 524288
  const int nw8 = DMODEL * DMODEL / 8;   // 131072
  cast_f32_to_bf16<<<(nx8 + 255) / 256, 256, 0, stream>>>(x,  xbf,  nx8);
  cast_f32_to_bf16<<<(nw8 + 255) / 256, 256, 0, stream>>>(Wq, wqbf, nw8);
  cast_f32_to_bf16<<<(nw8 + 255) / 256, 256, 0, stream>>>(Wk, wkbf, nw8);
  cast_f32_to_bf16<<<(nw8 + 255) / 256, 256, 0, stream>>>(Wv, wvbf, nw8);
  cast_f32_to_bf16<<<(nw8 + 255) / 256, 256, 0, stream>>>(Wo, wobf, nw8);

  // (ROWS/128) * (DMODEL/64) = 32 * 16 = 512 blocks of 4 waves.
  const int gemm_blocks = (ROWS / 128) * (DMODEL / 64);
  // log2(e) / sqrt(1024): base-2 softmax, scale folded into Q.
  const float qscale = 0.04508422f;
  gemm_xwT<1><<<gemm_blocks, 128, 0, stream>>>(xbf, wqbf, bq, qbf, qscale);
  gemm_xwT<1><<<gemm_blocks, 128, 0, stream>>>(xbf, wkbf, bk, kbf, 1.0f);
  gemm_xwT<2><<<gemm_blocks, 128, 0, stream>>>(xbf, wvbf, bv, vtbf, 1.0f);

  // b*h*(SEQ/16) = 4096 waves; 4 waves / block
  attn_kernel<<<4096 / 4, 128, 0, stream>>>(qbf, kbf, vtbf, attnb);

  gemm_xwT<0><<<gemm_blocks, 128, 0, stream>>>(attnb, wobf, bo, d_out, 1.0f);
}
